// MultiHeadAttention_34351148434263
// MI455X (gfx1250) — compile-verified, matrix operand order checked by reading, not run on hardware
//
#include <hip/hip_runtime.h>
#include <hip/hip_bf16.h>

#define D_MODEL 1024
#define N_HEAD  16
#define BATCH   2
#define SEQ     2048
#define DH      64

typedef __attribute__((ext_vector_type(16))) __bf16 v16bf;
typedef __attribute__((ext_vector_type(8)))  float  v8f;
typedef __attribute__((ext_vector_type(4)))  int    i32x4;

#define AS_GLOBAL __attribute__((address_space(1)))
#define AS_LDS    __attribute__((address_space(3)))

// CDNA5 async global->LDS path (ASYNCcnt), if the toolchain exposes it.
#if defined(__has_builtin)
#  if __has_builtin(__builtin_amdgcn_global_load_async_to_lds_b128)
#    define HAVE_ASYNC_LDS 1
#  endif
#endif
#ifndef HAVE_ASYNC_LDS
#  define HAVE_ASYNC_LDS 0
#endif

union FragBF { v16bf v; uint4 u[2]; };

// Load one 16x32 bf16 WMMA operand fragment (A or B side of an NT GEMM) from
// row-major storage with leading dim `ldm` (elements). Per the CDNA5 16-bit
// operand layout: lanes 0-15 hold row (lane), K = 0..7 and 16..23; lanes 16-31
// hold the same rows, K = 8..15 and 24..31. Both chunks are contiguous 16B.
__device__ __forceinline__ v16bf load_frag(const __hip_bfloat16* __restrict__ base,
                                           int ldm, int row_base) {
  const int lane = threadIdx.x & 31;
  const int r  = row_base + (lane & 15);
  const int kb = (lane >> 4) * 8;
  FragBF f;
  f.u[0] = *reinterpret_cast<const uint4*>(base + (size_t)r * ldm + kb);
  f.u[1] = *reinterpret_cast<const uint4*>(base + (size_t)r * ldm + kb + 16);
  return f.v;
}

__device__ __forceinline__ v8f wmma_bf16(v16bf a, v16bf b, v8f c) {
  return __builtin_amdgcn_wmma_f32_16x16x32_bf16(false, a, false, b, (short)0, c,
                                                 false, false);
}

__device__ __forceinline__ float wred_max(float v) {
  for (int o = 16; o > 0; o >>= 1) v = fmaxf(v, __shfl_xor(v, o));
  return v;
}
__device__ __forceinline__ float wred_sum(float v) {
  for (int o = 16; o > 0; o >>= 1) v += __shfl_xor(v, o);
  return v;
}

// ---------------------------------------------------------------------------
// Projection GEMM: Out = X @ W^T + bias, X f32 [4096,1024], W f32 [1024,1024].
// MODE 0: write bf16 head-major  [b,h,s,dh]
// MODE 1: write bf16 head-major transposed [b,h,dh,s]   (for V)
// 128x128 block, 256 threads (8 waves), each wave a 32x64 sub-tile.
// ---------------------------------------------------------------------------
template <int MODE>
__global__ __launch_bounds__(256) void gemm_nt_proj(
    const float* __restrict__ X, const float* __restrict__ W,
    const float* __restrict__ bias, __hip_bfloat16* __restrict__ Obf) {
  __shared__ __hip_bfloat16 As[128][40];  // 32 cols + 8 pad (bank-friendly)
  __shared__ __hip_bfloat16 Bs[128][40];

  const int tid  = threadIdx.x;
  const int lane = tid & 31;
  const int w    = tid >> 5;
  const int wm   = w & 3;   // 4 waves along M
  const int wn   = w >> 2;  // 2 waves along N
  const int mBase = blockIdx.x * 128;
  const int nBase = blockIdx.y * 128;

  v8f acc[2][4];
  for (int t = 0; t < 2; ++t)
    for (int u = 0; u < 4; ++u)
      for (int e = 0; e < 8; ++e) acc[t][u][e] = 0.f;

  for (int k0 = 0; k0 < D_MODEL; k0 += 32) {
    // Stage + convert A (128x32 f32 -> bf16)
    for (int i = tid; i < 128 * 8; i += 256) {
      const int r = i >> 3, c = (i & 7) * 4;
      const float4 a4 =
          *reinterpret_cast<const float4*>(X + (size_t)(mBase + r) * D_MODEL + k0 + c);
      As[r][c + 0] = __float2bfloat16(a4.x);
      As[r][c + 1] = __float2bfloat16(a4.y);
      As[r][c + 2] = __float2bfloat16(a4.z);
      As[r][c + 3] = __float2bfloat16(a4.w);
    }
    // Stage + convert B (rows of W, 128x32)
    for (int i = tid; i < 128 * 8; i += 256) {
      const int r = i >> 3, c = (i & 7) * 4;
      const float4 b4 =
          *reinterpret_cast<const float4*>(W + (size_t)(nBase + r) * D_MODEL + k0 + c);
      Bs[r][c + 0] = __float2bfloat16(b4.x);
      Bs[r][c + 1] = __float2bfloat16(b4.y);
      Bs[r][c + 2] = __float2bfloat16(b4.z);
      Bs[r][c + 3] = __float2bfloat16(b4.w);
    }
    if (k0 + 32 < D_MODEL) {  // hint next k-tile into cache
      __builtin_prefetch(X + (size_t)(mBase + (tid >> 1)) * D_MODEL + k0 + 32, 0, 0);
      __builtin_prefetch(W + (size_t)(nBase + (tid >> 1)) * D_MODEL + k0 + 32, 0, 0);
    }
    __syncthreads();

    v16bf aF[2], bF[4];
    for (int t = 0; t < 2; ++t) aF[t] = load_frag(&As[0][0], 40, wm * 32 + t * 16);
    for (int u = 0; u < 4; ++u) bF[u] = load_frag(&Bs[0][0], 40, wn * 64 + u * 16);
    for (int t = 0; t < 2; ++t)
      for (int u = 0; u < 4; ++u) acc[t][u] = wmma_bf16(aF[t], bF[u], acc[t][u]);
    __syncthreads();
  }

  // Epilogue: C element (m = vgpr + 8*(lane>>4), n = lane&15) per 16x16 tile.
  const int mtop = (lane >> 4) * 8;
  const int ncol = lane & 15;
  for (int t = 0; t < 2; ++t)
    for (int u = 0; u < 4; ++u) {
      const int ng = nBase + wn * 64 + u * 16 + ncol;
      const float bv = bias[ng];
      const int h = ng >> 6, d = ng & 63;
      for (int j = 0; j < 8; ++j) {
        const int mg = mBase + wm * 32 + t * 16 + mtop + j;
        const int b = mg >> 11, s = mg & (SEQ - 1);
        const float val = acc[t][u][j] + bv;
        if (MODE == 0)
          Obf[((size_t)(b * N_HEAD + h) * SEQ + s) * DH + d] = __float2bfloat16(val);
        else
          Obf[((size_t)(b * N_HEAD + h) * DH + d) * SEQ + s] = __float2bfloat16(val);
      }
    }
}

// ---------------------------------------------------------------------------
// Output GEMM: Out = attn(bf16) @ Wo^T + bo -> f32 [4096,1024]
// A-tiles are already bf16, so they go memory->LDS through the CDNA5 async
// path (global_load_async_to_lds_b128, ASYNCcnt) with no VGPR round-trip,
// overlapping with the f32->bf16 conversion of the W tile.
// ---------------------------------------------------------------------------
__global__ __launch_bounds__(256) void gemm_nt_out(
    const __hip_bfloat16* __restrict__ A, const float* __restrict__ W,
    const float* __restrict__ bias, float* __restrict__ Out) {
  __shared__ __hip_bfloat16 As[128][40];
  __shared__ __hip_bfloat16 Bs[128][40];

  const int tid  = threadIdx.x;
  const int lane = tid & 31;
  const int w    = tid >> 5;
  const int wm   = w & 3;
  const int wn   = w >> 2;
  const int mBase = blockIdx.x * 128;
  const int nBase = blockIdx.y * 128;

  v8f acc[2][4];
  for (int t = 0; t < 2; ++t)
    for (int u = 0; u < 4; ++u)
      for (int e = 0; e < 8; ++e) acc[t][u][e] = 0.f;

  for (int k0 = 0; k0 < D_MODEL; k0 += 32) {
    // Stage A (bf16 128x32): async DMA into LDS when available.
    for (int i = tid; i < 128 * 4; i += 256) {
      const int r = i >> 2, c = (i & 3) * 8;
      const __hip_bfloat16* gp = A + (size_t)(mBase + r) * D_MODEL + k0 + c;
#if HAVE_ASYNC_LDS
      __builtin_amdgcn_global_load_async_to_lds_b128(
          (AS_GLOBAL i32x4*)gp, (AS_LDS i32x4*)&As[r][c], 0, 0);
#else
      *reinterpret_cast<uint4*>(&As[r][c]) =
          *reinterpret_cast<const uint4*>(gp);
#endif
    }
    // Stage + convert B (rows of Wo, 128x32 f32 -> bf16) while A streams in.
    for (int i = tid; i < 128 * 8; i += 256) {
      const int r = i >> 3, c = (i & 7) * 4;
      const float4 b4 =
          *reinterpret_cast<const float4*>(W + (size_t)(nBase + r) * D_MODEL + k0 + c);
      Bs[r][c + 0] = __float2bfloat16(b4.x);
      Bs[r][c + 1] = __float2bfloat16(b4.y);
      Bs[r][c + 2] = __float2bfloat16(b4.z);
      Bs[r][c + 3] = __float2bfloat16(b4.w);
    }
#if HAVE_ASYNC_LDS
    asm volatile("s_wait_asynccnt 0" ::: "memory");
#endif
    __syncthreads();

    v16bf aF[2], bF[4];
    for (int t = 0; t < 2; ++t) aF[t] = load_frag(&As[0][0], 40, wm * 32 + t * 16);
    for (int u = 0; u < 4; ++u) bF[u] = load_frag(&Bs[0][0], 40, wn * 64 + u * 16);
    for (int t = 0; t < 2; ++t)
      for (int u = 0; u < 4; ++u) acc[t][u] = wmma_bf16(aF[t], bF[u], acc[t][u]);
    __syncthreads();
  }

  const int mtop = (lane >> 4) * 8;
  const int ncol = lane & 15;
  for (int t = 0; t < 2; ++t)
    for (int u = 0; u < 4; ++u) {
      const int ng = nBase + wn * 64 + u * 16 + ncol;
      const float bv = bias[ng];
      for (int j = 0; j < 8; ++j) {
        const int mg = mBase + wm * 32 + t * 16 + mtop + j;
        Out[(size_t)mg * D_MODEL + ng] = acc[t][u][j] + bv;
      }
    }
}

// ---------------------------------------------------------------------------
// Attention with faithful double softmax. One block = one (b,h) and a 16-query
// tile. Full 16x2048 logits panel lives in LDS (f32), softmax in f32, scores
// re-packed to bf16 in LDS and fed back through WMMA against V^T.
// Second softmax row-max is analytic: max(p) = (1/Z1)/8.
// ---------------------------------------------------------------------------
#define SB_LD 2056  // padded bf16 score row (2048 + 8)

__global__ __launch_bounds__(256) void attention_double_softmax(
    const __hip_bfloat16* __restrict__ Qp, const __hip_bfloat16* __restrict__ Kp,
    const __hip_bfloat16* __restrict__ Vt, __hip_bfloat16* __restrict__ attn) {
  extern __shared__ char smem[];
  float* logits = reinterpret_cast<float*>(smem);                       // [16][2048] f32
  __hip_bfloat16* Sbuf =
      reinterpret_cast<__hip_bfloat16*>(smem + 16 * SEQ * 4);           // [16][SB_LD] bf16
  float* outAcc =
      reinterpret_cast<float*>(smem + 16 * SEQ * 4 + 16 * SB_LD * 2);   // [16][64] f32

  const int tid  = threadIdx.x;
  const int lane = tid & 31;
  const int w    = tid >> 5;
  const int qt   = blockIdx.x & 127;
  const int bh   = blockIdx.x >> 7;
  const int qbase = qt * 16;

  const __hip_bfloat16* Qh = Qp + ((size_t)bh * SEQ + qbase) * DH;
  const __hip_bfloat16* Kh = Kp + (size_t)bh * SEQ * DH;
  const __hip_bfloat16* Vh = Vt + (size_t)bh * DH * SEQ;  // [64][2048]

  // Q operand fragments (16 rows x 64 K, two k-steps), shared by all waves.
  v16bf qF0 = load_frag(Qh, DH, 0);
  v16bf qF1 = load_frag(Qh + 32, DH, 0);

  // ---- Phase 1: logits = Q @ K^T; wave w covers keys [w*256, w*256+256) ----
  for (int nt = 0; nt < 16; ++nt) {
    const int nbase = w * 256 + nt * 16;
    v8f acc;
    for (int e = 0; e < 8; ++e) acc[e] = 0.f;
    acc = wmma_bf16(qF0, load_frag(Kh, DH, nbase), acc);
    acc = wmma_bf16(qF1, load_frag(Kh + 32, DH, nbase), acc);
    const int n = nbase + (lane & 15);
    const int mtop = (lane >> 4) * 8;
    for (int j = 0; j < 8; ++j) logits[(mtop + j) * SEQ + n] = acc[j];
  }
  __syncthreads();

  // ---- Phase 2: double softmax, 2 rows per wave ----
  for (int rr = 0; rr < 2; ++rr) {
    const int row = w * 2 + rr;
    const float* lr = logits + (size_t)row * SEQ;
    float m1 = -3.4e38f;
    for (int i = lane; i < SEQ; i += 32) m1 = fmaxf(m1, lr[i]);
    m1 = wred_max(m1);
    float z1 = 0.f;
    for (int i = lane; i < SEQ; i += 32) z1 += __expf(lr[i] - m1);
    z1 = wred_sum(z1);
    const float inv = 1.f / (8.f * z1);  // p_i = exp(l-m1)*inv ; max(p) == inv
    float z2 = 0.f;
    for (int i = lane; i < SEQ; i += 32)
      z2 += __expf(__expf(lr[i] - m1) * inv - inv);
    z2 = wred_sum(z2);
    const float iz2 = 1.f / z2;
    __hip_bfloat16* srow = Sbuf + (size_t)row * SB_LD;
    for (int i = lane; i < SEQ; i += 32)
      srow[i] = __float2bfloat16(__expf(__expf(lr[i] - m1) * inv - inv) * iz2);
  }
  for (int i = tid; i < 16 * 64; i += 256) outAcc[i] = 0.f;
  __syncthreads();

  // ---- Phase 3: out(16x64) = S(16x2048) @ V ; V^T rows are B operand (NT) ----
  // 8 waves = 4 n-tiles x 2 K-halves; reduce partials via ds_add_f32.
  const int nt = w & 3;
  const int khalf = w >> 2;
  v8f acc;
  for (int e = 0; e < 8; ++e) acc[e] = 0.f;
  for (int kit = khalf * 32; kit < khalf * 32 + 32; ++kit) {
    const int k0 = kit * 32;
    v16bf aF = load_frag(Sbuf + k0, SB_LD, 0);
    v16bf bF = load_frag(Vh + k0, SEQ, nt * 16);
    acc = wmma_bf16(aF, bF, acc);
  }
  {
    const int n = nt * 16 + (lane & 15);
    const int mtop = (lane >> 4) * 8;
    for (int j = 0; j < 8; ++j) atomicAdd(&outAcc[(mtop + j) * 64 + n], acc[j]);
  }
  __syncthreads();

  const int b = bh >> 4, h = bh & 15;
  for (int i = tid; i < 16 * 64; i += 256) {
    const int m = i >> 6, n = i & 63;
    attn[((size_t)b * SEQ + qbase + m) * D_MODEL + h * DH + n] =
        __float2bfloat16(outAcc[i]);
  }
}

// ---------------------------------------------------------------------------
extern "C" void kernel_launch(void* const* d_in, const int* in_sizes, int n_in,
                              void* d_out, int out_size, void* d_ws, size_t ws_size,
                              hipStream_t stream) {
  const float* q  = (const float*)d_in[0];
  const float* k  = (const float*)d_in[1];
  const float* v  = (const float*)d_in[2];
  const float* Wq = (const float*)d_in[3];
  const float* bq = (const float*)d_in[4];
  const float* Wk = (const float*)d_in[5];
  const float* bk = (const float*)d_in[6];
  const float* Wv = (const float*)d_in[7];
  const float* bv = (const float*)d_in[8];
  const float* Wo = (const float*)d_in[9];
  const float* bo = (const float*)d_in[10];
  float* out = (float*)d_out;

  char* ws = (char*)d_ws;
  __hip_bfloat16* Qp   = (__hip_bfloat16*)(ws);                       // 8 MB
  __hip_bfloat16* Kp   = (__hip_bfloat16*)(ws + (size_t)(8 << 20));   // 8 MB
  __hip_bfloat16* Vt   = (__hip_bfloat16*)(ws + (size_t)(16 << 20));  // 8 MB
  __hip_bfloat16* attn = (__hip_bfloat16*)(ws + (size_t)(24 << 20));  // 8 MB

  dim3 grid(32, 8), blk(256);
  gemm_nt_proj<0><<<grid, blk, 0, stream>>>(q, Wq, bq, Qp);
  gemm_nt_proj<0><<<grid, blk, 0, stream>>>(k, Wk, bk, Kp);
  gemm_nt_proj<1><<<grid, blk, 0, stream>>>(v, Wv, bv, Vt);

  const size_t lds = (size_t)16 * SEQ * 4 + (size_t)16 * SB_LD * 2 + 16 * 64 * 4;
  attention_double_softmax<<<dim3(BATCH * N_HEAD * (SEQ / 16)), blk, lds, stream>>>(
      Qp, Kp, Vt, attn);

  gemm_nt_out<<<grid, blk, 0, stream>>>(attn, Wo, bo, out);
}
